// MixBlock_20315195310839
// MI455X (gfx1250) — matrix operand order repaired and checked by reading, not compiled, not run on hardware
//
#include <hip/hip_runtime.h>
#include <hip/hip_bf16.h>

// ---------- types ----------
typedef __bf16 bf16_t;
typedef bf16_t v16bf __attribute__((ext_vector_type(16)));
typedef float  v8f   __attribute__((ext_vector_type(8)));
typedef unsigned short u16x8  __attribute__((ext_vector_type(8)));
typedef unsigned short u16x16 __attribute__((ext_vector_type(16)));
typedef unsigned int   u32x4  __attribute__((ext_vector_type(4)));
typedef unsigned int   u32x8  __attribute__((ext_vector_type(8)));

// Problem constants: B=16, H=W=64, C=256
#define NB   16
#define NH   64
#define NW   64
#define NC   256
#define NPIX (NB * NH * NW)               // 65536
#define QK_ELEMS (NB * NW * 2 * NH * NC)  // 33,554,432 bf16 each for Q and KT
#define ATT_ELEMS (NB * NC * NW * NW)     // 16,777,216 bf16

__device__ __forceinline__ unsigned short f2bf(float f) {
  unsigned int u = __builtin_bit_cast(unsigned int, f);
  u += 0x7FFFu + ((u >> 16) & 1u);        // round-to-nearest-even
  return (unsigned short)(u >> 16);
}
__device__ __forceinline__ float bf2f(unsigned short h) {
  unsigned int u = ((unsigned int)h) << 16;
  return __builtin_bit_cast(float, u);
}
// Build a v16bf WMMA fragment from two 8x16-bit LDS runs (each 16B aligned).
__device__ __forceinline__ v16bf ldsfrag(const unsigned short* p0,
                                         const unsigned short* p1) {
  u16x8 a = *(const u16x8*)p0;
  u16x8 b = *(const u16x8*)p1;
  u16x16 c = __builtin_shufflevector(a, b, 0,1,2,3,4,5,6,7,8,9,10,11,12,13,14,15);
  return __builtin_bit_cast(v16bf, c);
}

// ---------------- TDM: tensor_load_to_lds via inline asm --------------------
// Issues one Tensor-Data-Mover DMA (global -> LDS) described by a 2-group D#.
// group0: count=1, lds_addr, global_addr[56:0], type=2.  group1: dims/pads.
// Tracked with TENSORcnt; caller waits with s_wait_tensorcnt.
__device__ __forceinline__ void tdm_load(unsigned lds_off, const void* gptr,
                                         unsigned w0, unsigned w1, unsigned w2,
                                         unsigned w3, unsigned w4, unsigned w5) {
  unsigned long long ga = (unsigned long long)gptr;
  u32x4 g0;
  g0[0] = 1u;                                            // count=1 (valid D#)
  g0[1] = lds_off;                                       // LDS byte address
  g0[2] = (unsigned)ga;                                  // global_addr[31:0]
  g0[3] = ((unsigned)(ga >> 32) & 0x01FFFFFFu)           // global_addr[56:32]
          | 0x80000000u;                                 // type=2 ("image")
  u32x8 g1;
  g1[0] = w0; g1[1] = w1; g1[2] = w2; g1[3] = w3;
  g1[4] = w4; g1[5] = w5; g1[6] = 0u; g1[7] = 0u;
  asm volatile("tensor_load_to_lds %0, %1" :: "s"(g0), "s"(g1) : "memory");
}

// D# group1 for: 1-D copy of 4096 DWORDs, +4 DW LDS pad every 64 DW
// (=> 136-ushort padded rows for a 64 x 128 bf16 tile).
// data_size=4B(code2), pad_en, pad_interval=64DW(code5), pad_amount=4DW(code3)
#define TDM1D_W0 ((2u << 16) | (1u << 20) | (5u << 22) | (3u << 25))
#define TDM1D_W1 (4096u << 16)          // tensor_dim0 = 4096 (low 16 -> hi half)
#define TDM1D_W2 (1u << 16)             // tensor_dim1 = 1
#define TDM1D_W3 (4096u << 16)          // tile_dim0 = 4096
#define TDM1D_W4 (0u)                   // tile_dim1/2 unused
#define TDM1D_W5 (4096u)                // tensor_dim0_stride = 4096 DW

// D# group1 for: 2-D tile 16 DW x 256 rows from a 128-DW-stride tensor,
// +4 DW LDS pad per row (=> 40-ushort padded rows for 32 bf16 x 256).
// pad_interval=16DW(code3), pad_amount=4DW(code3)
#define TDMW_W0 ((2u << 16) | (1u << 20) | (3u << 22) | (3u << 25))
#define TDMW_W1 (128u << 16)            // tensor_dim0 = 128 DW (full row)
#define TDMW_W2 (256u << 16)            // tensor_dim1 = 256 rows
#define TDMW_W3 (16u << 16)             // tile_dim0 = 16 DW (64B slice)
#define TDMW_W4 (256u)                  // tile_dim1 = 256
#define TDMW_W5 (128u)                  // tensor_dim0_stride = 128 DW

// ---------------- Kernel 0: weight transpose + bf16 convert -----------------
// Wt[mt][n][k] = W_mt[k][n] (bf16), mt = {0:qf, 1:kf, 2:ql, 3:kl}
__global__ __launch_bounds__(256)
void wt_conv_kernel(const float* __restrict__ qf, const float* __restrict__ kf,
                    const float* __restrict__ ql, const float* __restrict__ kl,
                    unsigned short* __restrict__ Wt) {
  const float* W = (blockIdx.y == 0) ? qf : (blockIdx.y == 1) ? kf
                 : (blockIdx.y == 2) ? ql : kl;
  unsigned short* dst = Wt + (size_t)blockIdx.y * (NC * NC);
  const int k = threadIdx.x;
  const int nb = blockIdx.x;
  #pragma unroll
  for (int i = 0; i < 16; ++i) {
    int n = nb * 16 + i;
    dst[n * NC + k] = f2bf(W[k * NC + n]);
  }
}

// ---------------- Kernel 1: pointwise convs (bf16 WMMA GEMM) ----------------
// grid.x = 2048 tiles of 32 pixels, grid.y = input group (fad/lfs).
// Scatters bf16 results into Qb [B][W][2H][C] (reference reshape layout) and
// KTb, where KTb[n][j][k] is the per-batch pre-transposed K for stage 2.
__global__ __launch_bounds__(256)
void pw_conv_kernel(const float* __restrict__ x_fad, const float* __restrict__ x_lfs,
                    const float* __restrict__ qf_b, const float* __restrict__ ql_b,
                    const float* __restrict__ kf_b, const float* __restrict__ kl_b,
                    const unsigned short* __restrict__ Wt,
                    unsigned short* __restrict__ Qb, unsigned short* __restrict__ KTb) {
  __shared__ unsigned short sA[32 * 264];        // 32 pixels x 256 K (+8 pad)
  __shared__ unsigned short sW[2][256 * 40];     // 256 n x 32 k (+8 pad), q & k

  const int tid = threadIdx.x;
  const int g   = blockIdx.y;
  const int t   = blockIdx.x;
  const int b   = t >> 7;
  const int h   = (t >> 1) & 63;
  const int w0  = (t & 1) * 32;

  const float* x = g ? x_lfs : x_fad;
  const unsigned short* WtQ = Wt + (size_t)(g * 2 + 0) * (NC * NC);
  const unsigned short* WtK = Wt + (size_t)(g * 2 + 1) * (NC * NC);
  const float* biasQ = g ? ql_b : qf_b;
  const float* biasK = g ? kl_b : kf_b;

  // Stage A tile: 32 pixels x 256 channels, f32 -> bf16 LDS
  {
    const float4* xs = (const float4*)(x + ((size_t)(b * NH + h) * NW + w0) * NC);
    #pragma unroll
    for (int i = 0; i < 8; ++i) {
      int ch  = tid + i * 256;
      int row = ch >> 6;
      int c4  = ch & 63;
      float4 v = xs[row * 64 + c4];
      unsigned int lo = (unsigned int)f2bf(v.x) | ((unsigned int)f2bf(v.y) << 16);
      unsigned int hi = (unsigned int)f2bf(v.z) | ((unsigned int)f2bf(v.w) << 16);
      *(uint2*)&sA[row * 264 + c4 * 4] = make_uint2(lo, hi);
    }
  }

  const int lane = tid & 31;
  const int wv   = tid >> 5;
  const int m    = lane & 15;
  const int half = lane >> 4;
  const int rb   = wv >> 2;                      // 0..1
  const int cb0  = wv & 3;                       // cb = cb0 + 4*cc

  const unsigned sW0off = (unsigned)(unsigned long long)&sW[0][0];
  const unsigned sW1off = (unsigned)(unsigned long long)&sW[1][0];

  v8f acc[2][4] = {};

  for (int kc = 0; kc < 8; ++kc) {
    __syncthreads();                             // previous chunk's readers done
    if (tid < 32) {                              // one wave drives the TDM
      tdm_load(sW0off, WtQ + kc * 32,            // 16DWx256 tile, row stride 128DW
               TDMW_W0, TDMW_W1, TDMW_W2, TDMW_W3, TDMW_W4, TDMW_W5);
      tdm_load(sW1off, WtK + kc * 32,
               TDMW_W0, TDMW_W1, TDMW_W2, TDMW_W3, TDMW_W4, TDMW_W5);
      __builtin_amdgcn_s_wait_tensorcnt(0);
    }
    __syncthreads();                             // DMA data visible to all waves

    const unsigned short* arow = &sA[(rb * 16 + m) * 264 + kc * 32 + half * 8];
    v16bf af = ldsfrag(arow, arow + 16);
    #pragma unroll
    for (int mm = 0; mm < 2; ++mm) {
      #pragma unroll
      for (int cc = 0; cc < 4; ++cc) {
        int cb = cb0 + cc * 4;
        const unsigned short* brow = &sW[mm][(cb * 16 + m) * 40 + half * 8];
        v16bf bfrag = ldsfrag(brow, brow + 16);
        acc[mm][cc] = __builtin_amdgcn_wmma_f32_16x16x32_bf16(
            false, af, false, bfrag, (short)0, acc[mm][cc], false, false);
      }
    }
  }

  // Epilogue: bias + bf16 scatter
  const int h2 = g * NH + h;
  const int nbase = b * NC + h2 * 2 + (w0 >> 5); // n = b*256 + h2*2 + (w>>5)
  #pragma unroll
  for (int cc = 0; cc < 4; ++cc) {
    int cb = cb0 + cc * 4;
    int d  = cb * 16 + m;
    float bq = biasQ[d];
    float bk = biasK[d];
    #pragma unroll
    for (int r = 0; r < 8; ++r) {
      int wl = rb * 16 + half * 8 + r;           // w & 31
      int w  = w0 + wl;
      size_t qi = ((size_t)(b * NW + w) * (2 * NH) + h2) * NC + d;
      size_t ki = (size_t)nbase * 8192 + (size_t)(d & 63) * 128 + wl * 4 + (d >> 6);
      Qb[qi]  = f2bf(acc[0][cc][r] + bq);
      KTb[ki] = f2bf(acc[1][cc][r] + bk);
    }
  }
}

// --------- Kernel 2: batched energy GEMM + softmax -> bf16 attention --------
// One block per n in [0,4096): E = Q2[n](64x128) @ K2[n](128x64), softmax rows.
// Q and pre-transposed K tiles are contiguous 16KB blocks -> TDM loads with
// LDS padding (4 DW every 64 DW) producing conflict-free 136-ushort rows.
__global__ __launch_bounds__(256)
void attn_kernel(const unsigned short* __restrict__ Qb,
                 const unsigned short* __restrict__ KTb,
                 unsigned short* __restrict__ attb) {
  __shared__ unsigned short sQ[64 * 136];        // 64 x 128 (+8 pad)
  __shared__ unsigned short sKt[64 * 136];       // [j][k] (+8 pad)
  __shared__ float sE[64 * 68];                  // energy (+4 pad)

  const int tid = threadIdx.x;
  const int n   = blockIdx.x;

  if (tid < 32) {
    tdm_load((unsigned)(unsigned long long)&sQ[0],  Qb  + (size_t)n * 8192,
             TDM1D_W0, TDM1D_W1, TDM1D_W2, TDM1D_W3, TDM1D_W4, TDM1D_W5);
    tdm_load((unsigned)(unsigned long long)&sKt[0], KTb + (size_t)n * 8192,
             TDM1D_W0, TDM1D_W1, TDM1D_W2, TDM1D_W3, TDM1D_W4, TDM1D_W5);
    __builtin_amdgcn_s_wait_tensorcnt(0);
  }
  __syncthreads();

  const int lane = tid & 31;
  const int wv   = tid >> 5;
  const int m    = lane & 15;
  const int half = lane >> 4;
  const int rb   = wv >> 1;                      // 0..3
  const int cbb  = (wv & 1) * 2;

  v8f acc[2] = {};
  #pragma unroll
  for (int kc = 0; kc < 4; ++kc) {
    const unsigned short* arow = &sQ[(rb * 16 + m) * 136 + kc * 32 + half * 8];
    v16bf af = ldsfrag(arow, arow + 16);
    #pragma unroll
    for (int tt = 0; tt < 2; ++tt) {
      int cb = cbb + tt;
      const unsigned short* brow = &sKt[(cb * 16 + m) * 136 + kc * 32 + half * 8];
      v16bf bfrag = ldsfrag(brow, brow + 16);
      acc[tt] = __builtin_amdgcn_wmma_f32_16x16x32_bf16(
          false, af, false, bfrag, (short)0, acc[tt], false, false);
    }
  }
  #pragma unroll
  for (int tt = 0; tt < 2; ++tt) {
    int cb = cbb + tt;
    #pragma unroll
    for (int r = 0; r < 8; ++r) {
      int row = rb * 16 + half * 8 + r;
      sE[row * 68 + cb * 16 + m] = acc[tt][r];
    }
  }
  __syncthreads();

  if (tid < 64) {                                // row softmax + bf16 store
    const float* e = &sE[tid * 68];
    float mx = e[0];
    #pragma unroll 8
    for (int j = 1; j < 64; ++j) mx = fmaxf(mx, e[j]);
    float s = 0.0f;
    #pragma unroll 8
    for (int j = 0; j < 64; ++j) s += __expf(e[j] - mx);
    float inv = 1.0f / s;
    unsigned int* dst = (unsigned int*)(attb + (size_t)n * 4096 + tid * 64);
    #pragma unroll 8
    for (int j = 0; j < 32; ++j) {
      float a0 = __expf(e[2 * j] - mx) * inv;
      float a1 = __expf(e[2 * j + 1] - mx) * inv;
      dst[j] = (unsigned int)f2bf(a0) | ((unsigned int)f2bf(a1) << 16);
    }
  }
}

// ------- Kernel 3: gates + depthwise scale + BN + residual (fused) ----------
__global__ __launch_bounds__(256)
void mix_out_kernel(const float* __restrict__ x_fad, const float* __restrict__ x_lfs,
                    const unsigned short* __restrict__ attb,
                    const float* __restrict__ fad_dw_w, const float* __restrict__ fad_dw_b,
                    const float* __restrict__ lfs_dw_w, const float* __restrict__ lfs_dw_b,
                    const float* __restrict__ fbn_g, const float* __restrict__ fbn_b,
                    const float* __restrict__ fbn_m, const float* __restrict__ fbn_v,
                    const float* __restrict__ lbn_g, const float* __restrict__ lbn_b,
                    const float* __restrict__ lbn_m, const float* __restrict__ lbn_v,
                    const float* __restrict__ fad_gamma, const float* __restrict__ lfs_gamma,
                    float* __restrict__ y_fad, float* __restrict__ y_lfs) {
  __shared__ unsigned short attL[256 * 72];      // [c][j] (+8 pad)
  const int c  = threadIdx.x;
  const int bi = blockIdx.x;
  const int b  = bi >> 6;
  const int i  = bi & 63;

  const float dwfw = fad_dw_w[c], dwfb = fad_dw_b[c];
  const float dwlw = lfs_dw_w[c], dwlb = lfs_dw_b[c];
  const float sf  = fbn_g[c] * rsqrtf(fbn_v[c] + 1e-3f);
  const float bfc = fbn_b[c] - fbn_m[c] * sf;
  const float sl  = lbn_g[c] * rsqrtf(lbn_v[c] + 1e-3f);
  const float blc = lbn_b[c] - lbn_m[c] * sl;
  const float fg  = 2.0f / (1.0f + __expf(-fad_gamma[0])) - 1.0f;
  const float lg  = 2.0f / (1.0f + __expf(-lfs_gamma[0])) - 1.0f;

  const unsigned short* asrc = attb + (size_t)(b * NC + c) * 4096 + (size_t)i * 64;
  #pragma unroll
  for (int q = 0; q < 8; ++q)
    *(uint4*)&attL[c * 72 + q * 8] = *(const uint4*)(asrc + q * 8);
  __syncthreads();

  const size_t base = ((size_t)(b * NH + i) * NW) * NC + c;
  #pragma unroll 4
  for (int j = 0; j < 64; ++j) {
    float att = bf2f(attL[c * 72 + j]);
    size_t p  = base + (size_t)j * NC;
    float xf = x_fad[p];
    float xl = x_lfs[p];
    float fd = xl * att * lg * dwfw + dwfb;
    float ld = xf * att * fg * dwlw + dwlb;
    y_fad[p] = xf + fd * sf + bfc;
    y_lfs[p] = xl + ld * sl + blc;
  }
}

// ------------------------------- launcher -----------------------------------
extern "C" void kernel_launch(void* const* d_in, const int* in_sizes, int n_in,
                              void* d_out, int out_size, void* d_ws, size_t ws_size,
                              hipStream_t stream) {
  const float* x_fad = (const float*)d_in[0];
  const float* x_lfs = (const float*)d_in[1];
  const float* qf_w  = (const float*)d_in[2];
  const float* qf_b  = (const float*)d_in[3];
  const float* ql_w  = (const float*)d_in[4];
  const float* ql_b  = (const float*)d_in[5];
  const float* kf_w  = (const float*)d_in[6];
  const float* kf_b  = (const float*)d_in[7];
  const float* kl_w  = (const float*)d_in[8];
  const float* kl_b  = (const float*)d_in[9];
  const float* fad_dw_w = (const float*)d_in[10];
  const float* fad_dw_b = (const float*)d_in[11];
  const float* lfs_dw_w = (const float*)d_in[12];
  const float* lfs_dw_b = (const float*)d_in[13];
  const float* fbn_g = (const float*)d_in[14];
  const float* fbn_b = (const float*)d_in[15];
  const float* fbn_m = (const float*)d_in[16];
  const float* fbn_v = (const float*)d_in[17];
  const float* lbn_g = (const float*)d_in[18];
  const float* lbn_b = (const float*)d_in[19];
  const float* lbn_m = (const float*)d_in[20];
  const float* lbn_v = (const float*)d_in[21];
  const float* fad_gamma = (const float*)d_in[22];
  const float* lfs_gamma = (const float*)d_in[23];

  unsigned short* Qb   = (unsigned short*)d_ws;
  unsigned short* KTb  = Qb + QK_ELEMS;
  unsigned short* attb = KTb + QK_ELEMS;
  unsigned short* Wt   = attb + ATT_ELEMS;

  float* y_fad = (float*)d_out;
  float* y_lfs = y_fad + (size_t)NPIX * NC;

  wt_conv_kernel<<<dim3(16, 4), 256, 0, stream>>>(qf_w, kf_w, ql_w, kl_w, Wt);
  pw_conv_kernel<<<dim3(2048, 2), 256, 0, stream>>>(
      x_fad, x_lfs, qf_b, ql_b, kf_b, kl_b, Wt, Qb, KTb);
  attn_kernel<<<dim3(4096), 256, 0, stream>>>(Qb, KTb, attb);
  mix_out_kernel<<<dim3(1024), 256, 0, stream>>>(
      x_fad, x_lfs, attb,
      fad_dw_w, fad_dw_b, lfs_dw_w, lfs_dw_b,
      fbn_g, fbn_b, fbn_m, fbn_v,
      lbn_g, lbn_b, lbn_m, lbn_v,
      fad_gamma, lfs_gamma, y_fad, y_lfs);
}
